// RLCCMemory_57346403336648
// MI455X (gfx1250) — matrix operand order repaired and checked by reading, not compiled, not run on hardware
//
#include <hip/hip_runtime.h>
#include <hip/hip_bf16.h>
#include <hip/hip_fp16.h>

typedef __attribute__((ext_vector_type(16))) _Float16 v16h;
typedef __attribute__((ext_vector_type(8)))  _Float16 v8h;
typedef __attribute__((ext_vector_type(8)))  float    v8f;
typedef __attribute__((ext_vector_type(16))) float    v16f;

#define BATCH 256
#define DIMK  256
#define MEMN  65536
#define CLSN  8192
#define TEMP_F 0.05f
#define EPS_F  1e-6f

#define NF    128                 // features per block (N-tile)
#define NTSUB (NF / 16)           // 8 N-subtiles of 16
#define SROW_STRIDE 40            // LDS row stride in halves (80 B -> 20 banks, conflict-free)

// ---------------- workspace layout (bytes) ----------------
// clsSum : [CLSN, BATCH] f32   @ 0          (8 MB)  -- zeroed each call
// nums   : [CLSN] i32          @ 8388608    (32 KB) -- zeroed each call
// lossb  : [BATCH] f32         @ 8421376    (1 KB)  -- zeroed each call
// xh     : [BATCH, DIMK] f16   @ 8422400    (128 KB)
#define OFF_NUMS  8388608
#define OFF_LOSSB 8421376
#define OFF_XH    8422400
#define ZERO_WORDS ((OFF_LOSSB + BATCH*4) / 4)

__global__ void zero_kernel(int* __restrict__ p, int n) {
    int i = blockIdx.x * blockDim.x + threadIdx.x;
    if (i < n) p[i] = 0;
}

// One block per batch row: L2-normalize, fold in 1/TEMP, convert to f16.
__global__ void normalize_kernel(const float* __restrict__ in, _Float16* __restrict__ xh) {
    __shared__ float s[256];
    int b = blockIdx.x, t = threadIdx.x;
    float v = in[b * DIMK + t];
    s[t] = v * v;
    __syncthreads();
    for (int o = 128; o > 0; o >>= 1) {
        if (t < o) s[t] += s[t + o];
        __syncthreads();
    }
    float norm  = sqrtf(s[0]);
    float scale = 1.0f / (fmaxf(norm, 1e-12f) * TEMP_F);
    xh[b * DIMK + t] = (_Float16)(v * scale);
}

// Histogram of labels -> integer counts (exact, order-independent).
__global__ void count_kernel(const int* __restrict__ labels, int* __restrict__ nums) {
    int m = blockIdx.x * blockDim.x + threadIdx.x;
    if (m < MEMN) atomicAdd(&nums[labels[m]], 1);
}

// Fused GEMM (WMMA f16->f32) + label-segmented scatter-add into clsSum[C,B].
// Block = 256 threads = 8 waves; block handles NF=128 features x all 256 batch rows.
// Per K-chunk (32), the 256 threads cooperatively stage the 128x32 f32 feature slab
// into LDS as f16 (converted exactly once), then each wave runs 2 batch tiles x 8
// N-subtiles of v_wmma_f32_16x16x32_f16 with B fragments from LDS.
__global__ void __launch_bounds__(256)
gemm_segsum_kernel(const _Float16* __restrict__ xh,
                   const float*    __restrict__ feat,
                   const int*      __restrict__ labels,
                   float*          __restrict__ clsSum) {
    __shared__ alignas(64) _Float16 sB[NF * SROW_STRIDE];   // 10 KB

    const int t    = threadIdx.x;
    const int lane = t & 31;
    const int w    = t >> 5;
    const int l15  = lane & 15;
    const int hi   = lane >> 4;               // 0: lanes 0-15, 1: lanes 16-31
    const int nbase = blockIdx.x * NF;

    // staging assignment: thread t stages (row = t/2, K-half = t%2) of each 128x32 chunk
    const int srow  = t >> 1;                 // 0..127
    const int shalf = t & 1;                  // 0..1
    const float*    gsrc = feat + (size_t)(nbase + srow) * DIMK + shalf * 16;
    _Float16*       sdst = sB + srow * SROW_STRIDE + shalf * 16;

    // compute assignment: wave w owns batch tiles 2w and 2w+1
    const int bt0 = 2 * w, bt1 = 2 * w + 1;
    const _Float16* xp0 = xh + (size_t)(bt0 * 16 + l15) * DIMK;
    const _Float16* xp1 = xh + (size_t)(bt1 * 16 + l15) * DIMK;

    v8f acc0[NTSUB] = {};
    v8f acc1[NTSUB] = {};

    for (int kc = 0; kc < 8; ++kc) {
        const int kb = kc * 32;

        // ---- cooperative stage: 128x32 f32 -> f16 in LDS (one conversion total) ----
        v16f bf = *(const v16f*)(gsrc + kb);
        v8h h0, h1;
#pragma unroll
        for (int i = 0; i < 8; ++i) { h0[i] = (_Float16)bf[i]; h1[i] = (_Float16)bf[8 + i]; }
        *(v8h*)(sdst)     = h0;
        *(v8h*)(sdst + 8) = h1;
        __syncthreads();

        // ---- A fragments (16x32 f16): lane holds K [k0..k0+15] ----
        const int k0 = kb + hi * 16;
        v8h a0lo = *(const v8h*)(xp0 + k0);
        v8h a0hi = *(const v8h*)(xp0 + k0 + 8);
        v8h a1lo = *(const v8h*)(xp1 + k0);
        v8h a1hi = *(const v8h*)(xp1 + k0 + 8);
        v16h a0, a1;
#pragma unroll
        for (int i = 0; i < 8; ++i) {
            a0[i] = a0lo[i]; a0[8 + i] = a0hi[i];
            a1[i] = a1lo[i]; a1[8 + i] = a1hi[i];
        }

        // ---- 8 N-subtiles: B fragment from LDS, 2 WMMAs each ----
#pragma unroll
        for (int nt = 0; nt < NTSUB; ++nt) {
            const _Float16* bp = sB + (nt * 16 + l15) * SROW_STRIDE + hi * 16;
            v8h b0 = *(const v8h*)(bp);
            v8h b1 = *(const v8h*)(bp + 8);
            v16h bh;
#pragma unroll
            for (int i = 0; i < 8; ++i) { bh[i] = b0[i]; bh[8 + i] = b1[i]; }
            acc0[nt] = __builtin_amdgcn_wmma_f32_16x16x32_f16(false, a0, false, bh, (short)0, acc0[nt], false, false);
            acc1[nt] = __builtin_amdgcn_wmma_f32_16x16x32_f16(false, a1, false, bh, (short)0, acc1[nt], false, false);
        }
        __syncthreads();   // protect LDS before next chunk's stores
    }

    // D layout: VGPR r -> (batch = bt*16 + r + 8*hi, feature col = l15).
    // One label per (lane, nt); 8 consecutive-address f32 atomic adds per tile.
#pragma unroll
    for (int nt = 0; nt < NTSUB; ++nt) {
        const int m = nbase + nt * 16 + l15;
        const int c = labels[m];
        float* d0 = clsSum + (size_t)c * BATCH + bt0 * 16 + hi * 8;
        float* d1 = clsSum + (size_t)c * BATCH + bt1 * 16 + hi * 8;
#pragma unroll
        for (int r = 0; r < 8; ++r) {
            unsafeAtomicAdd(d0 + r, acc0[nt][r]);
            unsafeAtomicAdd(d1 + r, acc1[nt][r]);
        }
    }
}

// One block per batch row: masked softmax over classes + NLL for the target class.
__global__ void loss_kernel(const float* __restrict__ clsSum,
                            const int*   __restrict__ nums,
                            const int*   __restrict__ labels,
                            const int*   __restrict__ indexes,
                            float*       __restrict__ lossb) {
    __shared__ float ssum[256];
    __shared__ float stgt;
    const int b = blockIdx.x, t = threadIdx.x;
    const int tgt = labels[indexes[b]];
    if (t == 0) stgt = 0.0f;
    __syncthreads();

    float acc = 0.0f;
    for (int c = t; c < CLSN; c += 256) {
        int n = nums[c];
        if (n > 0) {
            float sim = clsSum[(size_t)c * BATCH + b] / (float)n;  // segment mean (scores already /TEMP)
            float e   = expf(sim);
            acc += e;
            if (c == tgt) stgt = e;   // unique writer
        }
    }
    ssum[t] = acc;
    __syncthreads();
    for (int o = 128; o > 0; o >>= 1) {
        if (t < o) ssum[t] += ssum[t + o];
        __syncthreads();
    }
    if (t == 0) {
        float masked = stgt / (ssum[0] + EPS_F);
        lossb[b] = -logf(masked + EPS_F);
    }
}

__global__ void finalize_kernel(const float* __restrict__ lossb, float* __restrict__ out) {
    __shared__ float s[256];
    int t = threadIdx.x;
    s[t] = lossb[t];
    __syncthreads();
    for (int o = 128; o > 0; o >>= 1) {
        if (t < o) s[t] += s[t + o];
        __syncthreads();
    }
    if (t == 0) out[0] = s[0] / (float)BATCH;
}

extern "C" void kernel_launch(void* const* d_in, const int* in_sizes, int n_in,
                              void* d_out, int out_size, void* d_ws, size_t ws_size,
                              hipStream_t stream) {
    const float* inputs   = (const float*)d_in[0];
    const float* features = (const float*)d_in[1];
    const int*   indexes  = (const int*)d_in[2];
    const int*   labels   = (const int*)d_in[3];
    float* out = (float*)d_out;

    char* ws = (char*)d_ws;
    float*    clsSum = (float*)ws;
    int*      nums   = (int*)(ws + OFF_NUMS);
    float*    lossb  = (float*)(ws + OFF_LOSSB);
    _Float16* xh     = (_Float16*)(ws + OFF_XH);

    zero_kernel<<<(ZERO_WORDS + 255) / 256, 256, 0, stream>>>((int*)ws, ZERO_WORDS);
    normalize_kernel<<<BATCH, 256, 0, stream>>>(inputs, xh);
    count_kernel<<<MEMN / 256, 256, 0, stream>>>(labels, nums);
    gemm_segsum_kernel<<<MEMN / NF, 256, 0, stream>>>(xh, features, labels, clsSum);
    loss_kernel<<<BATCH, 256, 0, stream>>>(clsSum, nums, labels, indexes, lossb);
    finalize_kernel<<<1, 256, 0, stream>>>(lossb, out);
}
